// SymbolicCausalSelfAttentionALiBi_10969346474688
// MI455X (gfx1250) — matrix operand order repaired and checked by reading, not compiled, not run on hardware
//
#include <hip/hip_runtime.h>
#include <math.h>

// Problem constants (B=2, T=2048, C=1024, H=16, D=64)
#define B_ 2
#define T_ 2048
#define C_ 1024
#define H_ 16
#define D_ 64

typedef __attribute__((ext_vector_type(16))) __bf16 v16bf;
typedef __attribute__((ext_vector_type(8)))  float  v8f;
typedef __attribute__((ext_vector_type(4)))  int    v4i;

struct alignas(16) U128 { unsigned int w[4]; };
union Frag { U128 u[2]; v16bf v; };   // 32 bytes = 8 VGPRs per lane
union H8   { U128 u;    __bf16 h[8]; };

__device__ __forceinline__ v8f wmma_bf16(v16bf a, v16bf b, v8f c) {
  // D = A(16x32 bf16) x B(32x16 bf16) + C(16x16 f32)
  return __builtin_amdgcn_wmma_f32_16x16x32_bf16(
      /*neg_a=*/false, a, /*neg_b=*/false, b,
      /*c_mod=*/(short)0, c, /*reuse_a=*/false, /*reuse_b=*/false);
}

// A-matrix (16x32, bf16) fragment from a row-major [row][k] source.
// Lanes 0-15 hold M=lane, K[0:8]+K[16:24]; lanes 16-31 hold M=lane-16,
// K[8:16]+K[24:32]. Both runs are 8 contiguous bf16 -> two 16B loads.
__device__ __forceinline__ Frag load_a_rowmajor(const __bf16* src, int ld,
                                                int rowBase, int kbase, int lane) {
  Frag f;
  int m  = lane & 15;
  int kh = (lane >> 4) * 8;
  const __bf16* p = src + (size_t)(rowBase + m) * ld + kbase + kh;
  f.u[0] = *(const U128*)(p);
  f.u[1] = *(const U128*)(p + 16);
  return f;
}

// B-matrix (32x16, bf16) fragment where element [k][n] = w[n][k] (w row-major,
// i.e. the "N-major, K contiguous" weight layout W[out][in]).
// Lanes 0-15 hold N=lane, K[0:8]+K[16:24]; lanes 16-31 hold N=lane-16, K[8:16]+K[24:32].
__device__ __forceinline__ Frag load_b_nk(const __bf16* w, int ld,
                                          int nBase, int kbase, int lane) {
  Frag f;
  int n  = lane & 15;
  int kh = (lane >> 4) * 8;
  const __bf16* p = w + (size_t)(nBase + n) * ld + kbase + kh;
  f.u[0] = *(const U128*)(p);
  f.u[1] = *(const U128*)(p + 16);
  return f;
}

// A fragment where the logical [BT, C] row-major matrix is physically stored
// as [B, H, T, D] (attention output). 8-aligned k-runs never cross a head.
__device__ __forceinline__ Frag load_a_bhtd(const __bf16* Ob, int mBase, int kb, int lane) {
  int m  = mBase + (lane & 15);
  int bb = m >> 11;              // / T_
  int t  = m & (T_ - 1);
  int kh = (lane >> 4) * 8;
  int k0 = kb + kh;
  int k1 = kb + 16 + kh;
  Frag a;
  a.u[0] = *(const U128*)(Ob + ((size_t)(bb * H_ + (k0 >> 6)) * T_ + t) * D_ + (k0 & 63));
  a.u[1] = *(const U128*)(Ob + ((size_t)(bb * H_ + (k1 >> 6)) * T_ + t) * D_ + (k1 & 63));
  return a;
}

#if __has_builtin(__builtin_amdgcn_global_load_async_to_lds_b128)
#define HAVE_ASYNC_LDS 1
__device__ __forceinline__ void async_g2l_b128(void* lds, const void* g) {
  // Param types on this toolchain: (int4 AS1*, int4 AS3*, imm offset, imm cpol)
  __builtin_amdgcn_global_load_async_to_lds_b128(
      (__attribute__((address_space(1))) v4i*)g,
      (__attribute__((address_space(3))) v4i*)lds,
      /*imm offset=*/0, /*cpol=*/0);
}
#else
#define HAVE_ASYNC_LDS 0
#endif

__global__ void f32_to_bf16(const float* __restrict__ src, __bf16* __restrict__ dst, int n) {
  int i = blockIdx.x * blockDim.x + threadIdx.x;
  if (i < n) dst[i] = (__bf16)src[i];
}

// qkv = x @ W_attn^T ; scatter into Q/K/V [B,H,T,D] bf16.
// 32x32 register tile per wave (2 A-frags x 2 B-frags -> 4 WMMA per k-step),
// M=4096 (128 tiles32), N=3072 (96 tiles32), K=1024.
__global__ void __launch_bounds__(128) qkv_gemm(const __bf16* __restrict__ xb,
                                                const __bf16* __restrict__ Wab,
                                                __bf16* __restrict__ Qg,
                                                __bf16* __restrict__ Kg,
                                                __bf16* __restrict__ Vg) {
  int lane = threadIdx.x & 31;
  int wid  = blockIdx.x * 4 + (threadIdx.x >> 5);
  int tm = wid / 96;
  int tn = wid % 96;
  v8f acc[2][2] = {{{}, {}}, {{}, {}}};
  for (int kb = 0; kb < C_; kb += 32) {
    Frag a0 = load_a_rowmajor(xb, C_, tm * 32,      kb, lane);
    Frag a1 = load_a_rowmajor(xb, C_, tm * 32 + 16, kb, lane);
    Frag b0 = load_b_nk(Wab, C_, tn * 32,      kb, lane);
    Frag b1 = load_b_nk(Wab, C_, tn * 32 + 16, kb, lane);
    acc[0][0] = wmma_bf16(a0.v, b0.v, acc[0][0]);
    acc[0][1] = wmma_bf16(a0.v, b1.v, acc[0][1]);
    acc[1][0] = wmma_bf16(a1.v, b0.v, acc[1][0]);
    acc[1][1] = wmma_bf16(a1.v, b1.v, acc[1][1]);
  }
  // C/D layout: lane holds column n = lane&15; rows r (lanes<16) / r+8 (lanes>=16).
  int rowOff = (lane >> 4) << 3;
#pragma unroll
  for (int tj = 0; tj < 2; ++tj) {
    int n     = tn * 32 + tj * 16 + (lane & 15);
    int which = n >> 10;       // 0=Q 1=K 2=V
    int c     = n & (C_ - 1);
    int h = c >> 6, d = c & 63;
    __bf16* dst = (which == 0) ? Qg : (which == 1) ? Kg : Vg;
#pragma unroll
    for (int ti = 0; ti < 2; ++ti) {
#pragma unroll
      for (int i = 0; i < 8; ++i) {
        int bt = tm * 32 + ti * 16 + rowOff + i;
        int bb = bt >> 11;
        int t  = bt & (T_ - 1);
        dst[((size_t)(bb * H_ + h) * T_ + t) * D_ + d] = (__bf16)acc[ti][tj][i];
      }
    }
  }
}

// Flash-attention with ALiBi. Block = 4 waves = one (b,h) x 64 query rows.
__global__ void __launch_bounds__(128) attn_kernel(const __bf16* __restrict__ Qg,
                                                   const __bf16* __restrict__ Kg,
                                                   const __bf16* __restrict__ Vg,
                                                   __bf16* __restrict__ Og) {
  __shared__ __align__(16) __bf16 sK[32 * 64];      // K block   [key][d]
  __shared__ __align__(16) __bf16 sVt[64 * 32];     // V^T block [d][key]
  __shared__ __align__(16) __bf16 sP[4][16 * 32];   // per-wave P scratch [m][k]

  int tid  = threadIdx.x;
  int lane = tid & 31;
  int wave = tid >> 5;
  int qBlk = blockIdx.x & 31;            // T/64 = 32 query blocks
  int h    = (blockIdx.x >> 5) & (H_ - 1);
  int b    = blockIdx.x >> 9;
  int qbase = qBlk * 64;
  int wq    = qbase + wave * 16;

  const __bf16* Qbh = Qg + (size_t)(b * H_ + h) * T_ * D_;
  const __bf16* Kbh = Kg + (size_t)(b * H_ + h) * T_ * D_;
  const __bf16* Vbh = Vg + (size_t)(b * H_ + h) * T_ * D_;
  __bf16*       Obh = Og + (size_t)(b * H_ + h) * T_ * D_;

  float slope = exp2f(-0.5f * (float)(h + 1));   // ALiBi slopes for H=16
  const float scale = 0.125f;                    // 1/sqrt(64)

  Frag aQ0 = load_a_rowmajor(Qbh, D_, wq, 0,  lane);
  Frag aQ1 = load_a_rowmajor(Qbh, D_, wq, 32, lane);

  v8f o0 = {}, o1 = {}, o2 = {}, o3 = {};
  float mi[8], li[8];
#pragma unroll
  for (int i = 0; i < 8; ++i) { mi[i] = -__builtin_inff(); li[i] = 0.f; }

  int numKB = qbase / 32 + 2;   // key blocks with keyBase <= qbase+63
  for (int kb = 0; kb < numKB; ++kb) {
    int keyBase = kb * 32;
    // --- stage K (32x64 bf16 = 4KB contiguous) into LDS ---
#if HAVE_ASYNC_LDS
    {
      const char* gk = (const char*)(Kbh + (size_t)keyBase * D_);
      char* lk = (char*)sK;
      async_g2l_b128(lk + tid * 16,          gk + tid * 16);
      async_g2l_b128(lk + (tid + 128) * 16,  gk + (tid + 128) * 16);
    }
#else
    {
      const U128* src = (const U128*)(Kbh + (size_t)keyBase * D_);
      U128* dst = (U128*)sK;
      dst[tid]       = src[tid];
      dst[tid + 128] = src[tid + 128];
    }
#endif
    // --- stage V transposed into LDS (register bounce; async can't transpose) ---
    {
      int row = tid >> 2;
      int cg  = (tid & 3) * 16;
      const __bf16* vsrc = Vbh + (size_t)(keyBase + row) * D_ + cg;
      H8 va, vb;
      va.u = *(const U128*)vsrc;
      vb.u = *(const U128*)(vsrc + 8);
#pragma unroll
      for (int j = 0; j < 8; ++j) {
        sVt[(cg + j) * 32 + row]     = va.h[j];
        sVt[(cg + 8 + j) * 32 + row] = vb.h[j];
      }
    }
#if HAVE_ASYNC_LDS
    asm volatile("s_wait_asynccnt 0" ::: "memory");
#endif
    __syncthreads();

    if (keyBase <= wq + 15) {   // wave-uniform: EXEC stays all-ones for WMMA
      // ---- S = (Q K^T): two 16-key subtiles, K-dim = D = 64 (2 wmma each) ----
      v8f s0 = {}, s1 = {};
      {
        Frag k00 = load_b_nk(sK, D_, 0, 0, lane);   // [d][key] via sK[key][d]
        Frag k01 = load_b_nk(sK, D_, 0, 32, lane);
        s0 = wmma_bf16(aQ0.v, k00.v, s0);
        s0 = wmma_bf16(aQ1.v, k01.v, s0);
        Frag k10 = load_b_nk(sK, D_, 16, 0, lane);
        Frag k11 = load_b_nk(sK, D_, 16, 32, lane);
        s1 = wmma_bf16(aQ0.v, k10.v, s1);
        s1 = wmma_bf16(aQ1.v, k11.v, s1);
      }
      int rowOff = (lane >> 4) << 3;
      int colLo  = keyBase + (lane & 15);
      float p0[8], p1[8], alpha[8];
#pragma unroll
      for (int i = 0; i < 8; ++i) {
        int r = wq + rowOff + i;
        float b0 = (colLo <= r)      ? slope * (float)(colLo - r)      : -__builtin_inff();
        float b1 = (colLo + 16 <= r) ? slope * (float)(colLo + 16 - r) : -__builtin_inff();
        float v0 = s0[i] * scale + b0;
        float v1 = s1[i] * scale + b1;
        // row max across the 16 lanes holding this row (wave32 halves)
        float rm = fmaxf(v0, v1);
#pragma unroll
        for (int msk = 1; msk < 16; msk <<= 1)
          rm = fmaxf(rm, __shfl_xor(rm, msk, 32));
        float nm = fmaxf(mi[i], rm);
        alpha[i] = __expf(mi[i] - nm);
        mi[i] = nm;
        p0[i] = __expf(v0 - nm);
        p1[i] = __expf(v1 - nm);
        float rs = p0[i] + p1[i];
#pragma unroll
        for (int msk = 1; msk < 16; msk <<= 1)
          rs += __shfl_xor(rs, msk, 32);
        li[i] = li[i] * alpha[i] + rs;
      }
#pragma unroll
      for (int i = 0; i < 8; ++i) {
        o0[i] *= alpha[i]; o1[i] *= alpha[i];
        o2[i] *= alpha[i]; o3[i] *= alpha[i];
      }
      // ---- bounce P through LDS to get the A-fragment layout ----
      __bf16* ps = sP[wave];
#pragma unroll
      for (int i = 0; i < 8; ++i) {
        ps[(rowOff + i) * 32 + (lane & 15)]      = (__bf16)p0[i];
        ps[(rowOff + i) * 32 + 16 + (lane & 15)] = (__bf16)p1[i];
      }
      asm volatile("s_wait_dscnt 0" ::: "memory");   // in-wave LDS RAW
      Frag aP  = load_a_rowmajor(ps, 32, 0, 0, lane);
      Frag vf0 = load_b_nk(sVt, 32, 0,  0, lane);    // [key][d] via sVt[d][key]
      Frag vf1 = load_b_nk(sVt, 32, 16, 0, lane);
      Frag vf2 = load_b_nk(sVt, 32, 32, 0, lane);
      Frag vf3 = load_b_nk(sVt, 32, 48, 0, lane);
      o0 = wmma_bf16(aP.v, vf0.v, o0);
      o1 = wmma_bf16(aP.v, vf1.v, o1);
      o2 = wmma_bf16(aP.v, vf2.v, o2);
      o3 = wmma_bf16(aP.v, vf3.v, o3);
    }
    __syncthreads();
  }

  // ---- epilogue: O / l, store bf16 [B,H,T,D] ----
  int rowOff = (lane >> 4) << 3;
#pragma unroll
  for (int i = 0; i < 8; ++i) {
    float inv = 1.0f / li[i];
    int t = wq + rowOff + i;
    __bf16* od = Obh + (size_t)t * D_ + (lane & 15);
    od[0]  = (__bf16)(o0[i] * inv);
    od[16] = (__bf16)(o1[i] * inv);
    od[32] = (__bf16)(o2[i] * inv);
    od[48] = (__bf16)(o3[i] * inv);
  }
}

// y = attnOut @ W_proj^T, f32 out. attnOut stored [B,H,T,D]; fold to [BT, C].
// 32x32 register tile per wave; M=4096 (128 tiles32), N=1024 (32 tiles32).
__global__ void __launch_bounds__(128) proj_gemm(const __bf16* __restrict__ Ob,
                                                 const __bf16* __restrict__ Wpb,
                                                 float* __restrict__ out) {
  int lane = threadIdx.x & 31;
  int wid  = blockIdx.x * 4 + (threadIdx.x >> 5);
  int tm = wid >> 5;          // 128 M tiles32
  int tn = wid & 31;          // 32  N tiles32
  v8f acc[2][2] = {{{}, {}}, {{}, {}}};
  for (int kb = 0; kb < C_; kb += 32) {
    Frag a0 = load_a_bhtd(Ob, tm * 32,      kb, lane);
    Frag a1 = load_a_bhtd(Ob, tm * 32 + 16, kb, lane);
    Frag b0 = load_b_nk(Wpb, C_, tn * 32,      kb, lane);
    Frag b1 = load_b_nk(Wpb, C_, tn * 32 + 16, kb, lane);
    acc[0][0] = wmma_bf16(a0.v, b0.v, acc[0][0]);
    acc[0][1] = wmma_bf16(a0.v, b1.v, acc[0][1]);
    acc[1][0] = wmma_bf16(a1.v, b0.v, acc[1][0]);
    acc[1][1] = wmma_bf16(a1.v, b1.v, acc[1][1]);
  }
  int rowOff = (lane >> 4) << 3;
#pragma unroll
  for (int tj = 0; tj < 2; ++tj) {
    int n = tn * 32 + tj * 16 + (lane & 15);
#pragma unroll
    for (int ti = 0; ti < 2; ++ti) {
#pragma unroll
      for (int i = 0; i < 8; ++i) {
        int bt = tm * 32 + ti * 16 + rowOff + i;
        out[(size_t)bt * C_ + n] = acc[ti][tj][i];
      }
    }
  }
}

extern "C" void kernel_launch(void* const* d_in, const int* in_sizes, int n_in,
                              void* d_out, int out_size, void* d_ws, size_t ws_size,
                              hipStream_t stream) {
  const float* x  = (const float*)d_in[0];
  const float* Wa = (const float*)d_in[1];
  const float* Wp = (const float*)d_in[2];
  float* out = (float*)d_out;

  // Workspace layout (bf16), ~50.3 MB total
  __bf16* p   = (__bf16*)d_ws;
  __bf16* xb  = p; p += (size_t)B_ * T_ * C_;
  __bf16* Wab = p; p += (size_t)3 * C_ * C_;
  __bf16* Wpb = p; p += (size_t)C_ * C_;
  __bf16* Qg  = p; p += (size_t)B_ * T_ * C_;
  __bf16* Kg  = p; p += (size_t)B_ * T_ * C_;
  __bf16* Vg  = p; p += (size_t)B_ * T_ * C_;
  __bf16* Og  = p; p += (size_t)B_ * T_ * C_;

  const int nx  = B_ * T_ * C_;
  const int na  = 3 * C_ * C_;
  const int npj = C_ * C_;
  f32_to_bf16<<<(nx + 255) / 256, 256, 0, stream>>>(x, xb, nx);
  f32_to_bf16<<<(na + 255) / 256, 256, 0, stream>>>(Wa, Wab, na);
  f32_to_bf16<<<(npj + 255) / 256, 256, 0, stream>>>(Wp, Wpb, npj);

  // 128 x 96 tiles32, one 32x32 tile per wave, 4 waves per block
  qkv_gemm<<<3072, 128, 0, stream>>>(xb, Wab, Qg, Kg, Vg);
  // B*H*(T/64) = 1024 blocks
  attn_kernel<<<1024, 128, 0, stream>>>(Qg, Kg, Vg, Og);
  // 128 x 32 tiles32
  proj_gemm<<<1024, 128, 0, stream>>>(Og, Wpb, out);
}
